// MultiHeadAttention_12300786336013
// MI455X (gfx1250) — matrix operand order repaired and checked by reading, not compiled, not run on hardware
//
#include <hip/hip_runtime.h>

// ---------------------------------------------------------------- constants
constexpr int B_  = 8;
constexpr int S_  = 2048;
constexpr int D_  = 128;
constexpr int H_  = 8;
constexpr int DH_ = 16;     // head dim -> exactly one WMMA N dimension

typedef __attribute__((ext_vector_type(16))) __bf16 v16bf;
typedef __attribute__((ext_vector_type(8)))  float  v8f;
typedef int v4i __attribute__((vector_size(16)));   // int4, matches builtin proto
typedef unsigned int u32x4 __attribute__((ext_vector_type(4)));
typedef int          i32x4 __attribute__((ext_vector_type(4)));
typedef int          i32x8 __attribute__((ext_vector_type(8)));

// -------------------------------------------------- async-copy abstraction
#ifndef __has_builtin
#define __has_builtin(x) 0
#endif

#if defined(__AMDGCN__) && __has_builtin(__builtin_amdgcn_global_load_async_to_lds_b128)
#define HAVE_ASYNC 1
#else
#define HAVE_ASYNC 0
#endif

#if HAVE_ASYNC
#define CP16(src, dst)                                                        \
  __builtin_amdgcn_global_load_async_to_lds_b128(                             \
      (__attribute__((address_space(1))) v4i*)(src),                          \
      (__attribute__((address_space(3))) v4i*)(dst), 0, 0)
#if __has_builtin(__builtin_amdgcn_s_wait_asynccnt)
#define WAITA(n) __builtin_amdgcn_s_wait_asynccnt(n)
#else
#define WAITA(n) asm volatile("s_wait_asynccnt %0" ::"i"(n) : "memory")
#endif
#else
#define CP16(src, dst)                                                        \
  do { *(float4*)(dst) = *(const float4*)(src); } while (0)
#define WAITA(n) ((void)0)
#endif

// -------------------------------------------------- TDM (Tensor Data Mover)
#if defined(__AMDGCN__) && __has_builtin(__builtin_amdgcn_tensor_load_to_lds) && \
    __has_builtin(__builtin_amdgcn_s_wait_tensorcnt)
#define HAVE_TDM 1
#define WAITT(n) __builtin_amdgcn_s_wait_tensorcnt(n)
#else
#define HAVE_TDM 0
#define WAITT(n) ((void)0)
#endif

#if HAVE_TDM
// 2D tile load: tileD1 rows x tileD0 elements of 2-byte data, row stride
// `stride` elements, from gaddr into LDS at laddr. One TDM op per calling wave.
__device__ __forceinline__ void tdm_load_2d_b16(unsigned long long gaddr,
                                                unsigned laddr, unsigned tensorD0,
                                                unsigned tensorD1, unsigned stride,
                                                unsigned tileD0, unsigned tileD1) {
  // D# group0: count=1 | is_restore=0; lds_addr; global_addr[56:0]; type=2
  u32x4 g0 = {1u, laddr, (unsigned)gaddr,
              (unsigned)((gaddr >> 32) & 0x01FFFFFFu) | (2u << 30)};
  // D# group1 per CDNA5 ISA bit layout (data_size=1 -> 2 bytes)
  i32x8 g1 = {(int)(1u << 16),
              (int)((tensorD0 & 0xFFFFu) << 16),
              (int)((tensorD0 >> 16) | ((tensorD1 & 0xFFFFu) << 16)),
              (int)((tensorD1 >> 16) | (tileD0 << 16)),
              (int)(tileD1 & 0xFFFFu),
              (int)stride,          // tensor_dim0_stride[31:0]
              0, 0};
  i32x4 z4 = {0, 0, 0, 0};          // groups 2/3 unused (2D tensor)
  i32x8 z8 = {0, 0, 0, 0, 0, 0, 0, 0};
  __builtin_amdgcn_tensor_load_to_lds(g0, g1, z4, z4, z8, 0);
}
#endif

// ============================================================ projection
// out = X @ [W_q | W_mem], converted to bf16.
// Q (cols   0..127, scaled by dh^-0.5) -> Qb [b,h,S,16]
// K (cols 128..255)                    -> Kb [b,h,S,16]
// V (cols 256..383)                    -> Vt [b,h,16,S]  (pre-transposed)
// One wave = one 16x16 output tile, K=128 in 4 bf16 WMMAs.
// WS is a compile-time W stride so all weight loads are immediate offsets.
template <int WS>
__device__ __forceinline__ v8f proj_tile(const float* __restrict__ xrow,
                                         const float* __restrict__ wcol,
                                         int hf) {
  v8f acc = {};
  const float* wp = wcol + (size_t)(16 * hf) * WS;   // per-lane base
#pragma unroll
  for (int k0 = 0; k0 < D_; k0 += 32) {
    // A fragment 16x32 (bf16): lanes 0-15 hold K = k0+{0..15},
    // lanes 16-31 hold K = k0+{16..31}
    v16bf a;
    const float* xr = xrow + k0 + 16 * hf;
#pragma unroll
    for (int e = 0; e < 8; ++e) a[e] = (__bf16)xr[e];
#pragma unroll
    for (int e = 0; e < 8; ++e) a[8 + e] = (__bf16)xr[8 + e];

    // B fragment 32x16: lane = output column, element e -> K = k0 + e + 16*hf
    v16bf bfrag;
#pragma unroll
    for (int e = 0; e < 16; ++e) bfrag[e] = (__bf16)wp[(size_t)(k0 + e) * WS];

    acc = __builtin_amdgcn_wmma_f32_16x16x32_bf16(false, a, false, bfrag,
                                                  (short)0, acc, false, false);
  }
  return acc;
}

__global__ __launch_bounds__(256) void proj_qkv(
    const float* __restrict__ X, const float* __restrict__ Wq,
    const float* __restrict__ Wm, __bf16* __restrict__ Qb,
    __bf16* __restrict__ Kb, __bf16* __restrict__ Vt) {
  const int wave = threadIdx.x >> 5;
  const int lane = threadIdx.x & 31;
  const int m    = lane & 15;
  const int hf   = lane >> 4;
  const int tile = blockIdx.x * 8 + wave;   // 24576 tiles total
  const int mt   = tile / 24;
  const int nt   = tile % 24;
  const int row  = mt * 16 + m;             // A-frag row for this lane
  const int ncol = nt * 16 + m;             // B-frag column for this lane
  const int ntu  = __builtin_amdgcn_readfirstlane(nt);  // wave-uniform branch

  const float* xrow = X + (size_t)row * D_;
  v8f acc;
  if (ntu < 8) acc = proj_tile<D_>(xrow, Wq + ncol, hf);              // Q
  else         acc = proj_tile<2 * D_>(xrow, Wm + (ncol - D_), hf);   // K|V

  const float sc = (ntu < 8) ? 0.25f : 1.0f;  // fold dh^-0.5 into Q
  if (ntu < 16) {                              // Q or K: [b,h,S,16] layout
    __bf16* dst = (ntu < 8) ? Qb : Kb;
    const int c = (ntu < 8) ? ncol : (ncol - D_);
#pragma unroll
    for (int r = 0; r < 8; ++r) {
      const int rg = mt * 16 + r + 8 * hf;   // C-frag row = r + 8*half
      const int b  = rg / S_;
      const int s  = rg % S_;
      dst[((size_t)(b * H_ + c / DH_) * S_ + s) * DH_ + (c % DH_)] =
          (__bf16)(acc[r] * sc);
    }
  } else {                                     // V: transposed [b,h,16,S]
    const int c = ncol - 2 * D_;
#pragma unroll
    for (int r = 0; r < 8; ++r) {
      const int rg = mt * 16 + r + 8 * hf;
      const int b  = rg / S_;
      const int s  = rg % S_;
      Vt[((size_t)(b * H_ + c / DH_) * DH_ + (c % DH_)) * S_ + s] = (__bf16)acc[r];
    }
  }
}

// ============================================================ flash attention
// One block = 8 waves = 128 query rows of one (b,h). 64 keys per iteration;
// K staged via per-lane async global->LDS copies (ASYNCcnt), V^T tile staged
// via one TDM 2D descriptor per iteration (TENSORcnt); both double-buffered.
//
// Logits are computed TRANSPOSED (A = K tile, B = Q^T), so softmaxed P values
// sit in registers already in the exact A-fragment layout required by P@V.
__global__ __launch_bounds__(256) void flash_attn(
    const __bf16* __restrict__ Qb, const __bf16* __restrict__ Kb,
    const __bf16* __restrict__ Vt, const float* __restrict__ mask,
    float* __restrict__ out) {
  constexpr int KT  = 64;        // keys per iteration (4 x 16-key tiles)
  constexpr int NIT = S_ / KT;   // 32
  const int bh  = blockIdx.y;
  const int b   = bh / H_;
  const int hh  = bh % H_;
  const int q0  = blockIdx.x * 128;
  const int tid = threadIdx.x;
  const int wave = tid >> 5;
  const int lane = tid & 31;
  const int m    = lane & 15;
  const int hf   = lane >> 4;

  __shared__ __align__(16) __bf16 Ks[2][KT * DH_];   // [key][dh]   2 KB each
  __shared__ __align__(16) __bf16 Vs[2][DH_ * KT];   // [dh][key]   2 KB each

  const __bf16* Kg = Kb + (size_t)bh * S_ * DH_;
  const __bf16* Vg = Vt + (size_t)bh * DH_ * S_;

  // Q B-fragment (32x16, K-dim = dh padded 16->32): lane m (hf=0) carries the
  // full q-row vector, hf=1 lanes carry the zero pad.
  v16bf qb = {};
  if (hf == 0) {
    const __bf16* qp = Qb + ((size_t)bh * S_ + q0 + wave * 16 + m) * DH_;
#pragma unroll
    for (int e = 0; e < 16; ++e) qb[e] = qp[e];
  }

  float mold = -__builtin_inff();   // running max for q-row (lane) m
  float lrow = 0.0f;                // running sum for q-row (lane) m
  v8f oacc = {};                    // O: row r+8hf (q), col m (dh)

  auto stage = [&](int bufsel, int j0s) {
    if (tid < 128) {                  // 2 KB of K: contiguous, ASYNCcnt
      CP16(Kg + (size_t)j0s * DH_ + tid * 8, &Ks[bufsel][tid * 8]);
    }
#if HAVE_TDM
    else if (wave == 4) {             // V^T tile: 16 x 64 bf16 via TDM
      tdm_load_2d_b16(
          (unsigned long long)(Vg + j0s),
          (unsigned)(size_t)(__attribute__((address_space(3))) void*)&Vs[bufsel][0],
          (unsigned)S_, (unsigned)DH_, (unsigned)S_, (unsigned)KT, (unsigned)DH_);
    }
#else
    else {                            // fallback: per-lane async V copies
      const int i = tid - 128, r = i >> 3, sg = i & 7;
      CP16(Vg + (size_t)r * S_ + j0s + sg * 8, &Vs[bufsel][r * KT + sg * 8]);
    }
#endif
  };

  stage(0, 0);
  for (int it = 0; it < NIT; ++it) {
    const int buf = it & 1;
    if (it + 1 < NIT) { stage(buf ^ 1, (it + 1) * KT); WAITA(1); WAITT(1); }
    else              { WAITA(0); WAITT(0); }
    __syncthreads();

    // 4 transposed-logit tiles: cc[t] element r = logit[key t*16+r+8hf][qrow m]
    v8f cc[4];
#pragma unroll
    for (int t = 0; t < 4; ++t) {
      v16bf ka = {};   // A-frag: row = key, hf=0 elements e -> dh e, hf=1 pad
      const __bf16* kr = &Ks[buf][(t * 16 + m) * DH_];
#pragma unroll
      for (int e = 0; e < 16; ++e) if (hf == 0) ka[e] = kr[e];
      v8f z = {};
      cc[t] = __builtin_amdgcn_wmma_f32_16x16x32_bf16(false, ka, false, qb,
                                                      (short)0, z, false, false);
    }

    // additive mask, key = t*16 + r + 8hf
    const float* mp = mask + (size_t)b * S_ + it * KT + 8 * hf;
    float tmax = -__builtin_inff();
#pragma unroll
    for (int t = 0; t < 4; ++t) {
      const float4 m0 = *(const float4*)(mp + t * 16);
      const float4 m1 = *(const float4*)(mp + t * 16 + 4);
      const float mv[8] = {m0.x, m0.y, m0.z, m0.w, m1.x, m1.y, m1.z, m1.w};
#pragma unroll
      for (int r = 0; r < 8; ++r) {
        cc[t][r] += (1.0f - mv[r]) * -1e30f;
        tmax = fmaxf(tmax, cc[t][r]);
      }
    }
    // per-q-row softmax: in-register reduction + one cross-half shuffle
    tmax = fmaxf(tmax, __shfl_xor(tmax, 16, 32));
    const float mn   = fmaxf(mold, tmax);
    const float corr = __expf(mold - mn);
    mold = mn;

    float srow = 0.0f;
#pragma unroll
    for (int t = 0; t < 4; ++t)
#pragma unroll
      for (int r = 0; r < 8; ++r) {
        const float p = __expf(cc[t][r] - mn);
        cc[t][r] = p;
        srow += p;
      }
    srow += __shfl_xor(srow, 16, 32);
    lrow = lrow * corr + srow;

    // P is already in A-frag layout: element e<8 = tile(2t)[e], e>=8 = tile(2t+1)
    v16bf pa0, pa1;
#pragma unroll
    for (int e = 0; e < 8; ++e) {
      pa0[e]     = (__bf16)cc[0][e];
      pa0[8 + e] = (__bf16)cc[1][e];
      pa1[e]     = (__bf16)cc[2][e];
      pa1[8 + e] = (__bf16)cc[3][e];
    }

    // rescale O: broadcast corr from lane (qrow) into O's row-in-VGPR layout
#pragma unroll
    for (int r = 0; r < 8; ++r) oacc[r] *= __shfl(corr, r + 8 * hf, 32);

    // V B-frags (32x16): lane = dh column, elements = keys 8hf+{0..7}, 16+8hf+{0..7}
    v16bf vb0, vb1;
    const __bf16* vr = &Vs[buf][m * KT + 8 * hf];
#pragma unroll
    for (int e = 0; e < 16; ++e) { vb0[e] = vr[e + (e & 8)]; vb1[e] = vr[32 + e + (e & 8)]; }

    oacc = __builtin_amdgcn_wmma_f32_16x16x32_bf16(false, pa0, false, vb0,
                                                   (short)0, oacc, false, false);
    oacc = __builtin_amdgcn_wmma_f32_16x16x32_bf16(false, pa1, false, vb1,
                                                   (short)0, oacc, false, false);

    __syncthreads();   // all waves done reading buf before it is restaged
  }

#pragma unroll
  for (int r = 0; r < 8; ++r) {
    const float li  = __shfl(lrow, r + 8 * hf, 32);
    const int   row = q0 + wave * 16 + r + 8 * hf;
    out[((size_t)b * S_ + row) * D_ + hh * DH_ + m] = oacc[r] / li;
  }
}

// ============================================================ launch
extern "C" void kernel_launch(void* const* d_in, const int* in_sizes, int n_in,
                              void* d_out, int out_size, void* d_ws,
                              size_t ws_size, hipStream_t stream) {
  const float* X    = (const float*)d_in[0];
  const float* mask = (const float*)d_in[1];
  const float* Wq   = (const float*)d_in[2];
  const float* Wm   = (const float*)d_in[3];
  float* out        = (float*)d_out;

  const size_t per = (size_t)B_ * H_ * S_ * DH_;   // 2M bf16 elems = 4 MB
  __bf16* Qb = (__bf16*)d_ws;
  __bf16* Kb = Qb + per;
  __bf16* Vt = Kb + per;

  // 24576 wave-tiles / 8 waves per block
  proj_qkv<<<3072, 256, 0, stream>>>(X, Wq, Wm, Qb, Kb, Vt);
  flash_attn<<<dim3(S_ / 128, B_ * H_), 256, 0, stream>>>(Qb, Kb, Vt, mask, out);
}
